// MoChA_46368466927980
// MI455X (gfx1250) — matrix operand order repaired and checked by reading, not compiled, not run on hardware
//
#include <hip/hip_runtime.h>
#include <hip/hip_bf16.h>
#include <math.h>

// ---------------------------------------------------------------------------
// MoChA on MI455X (gfx1250, wave32, WMMA).
// B=32, T=4096, KD=QD=512, AD=128, window=8.
// ---------------------------------------------------------------------------

#define B_SZ   32
#define T_LEN  4096
#define KD_    512
#define AD_    128
#define NSEG   16
#define NEG_INF (-3.402823466e38f)

typedef __attribute__((ext_vector_type(16))) __bf16 v16bf;
typedef __attribute__((ext_vector_type(8)))  float  v8f;
typedef __attribute__((ext_vector_type(4)))  float  f4;

// ---------------------------------------------------------------------------
// Kernel 0a: per-batch query projection + weight-normed v_eff.
// grid = 64 blocks (mat[2] x b[32]), 128 threads (one per a).
// ---------------------------------------------------------------------------
__global__ __launch_bounds__(128) void mocha_prep_vq(
    const float* __restrict__ query,
    const float* __restrict__ wq_m, const float* __restrict__ b_m,
    const float* __restrict__ vw_m, const float* __restrict__ vg_m,
    const float* __restrict__ wq_c, const float* __restrict__ b_c,
    const float* __restrict__ vw_c, const float* __restrict__ vg_c,
    float* __restrict__ qb_m, float* __restrict__ qb_c,
    float* __restrict__ veff_m, float* __restrict__ veff_c)
{
    const int mat = blockIdx.x >> 5;       // 0 = mono, 1 = chunk
    const int b   = blockIdx.x & 31;
    const int a   = threadIdx.x;           // 0..127

    const float* vw   = mat ? vw_c : vw_m;
    const float  vg   = (mat ? vg_c : vg_m)[0];
    const float* wq   = mat ? wq_c : wq_m;
    const float* bias = mat ? b_c  : b_m;
    float*       qb   = mat ? qb_c : qb_m;
    float*       veff = mat ? veff_c : veff_m;

    // redundant per-thread norm over 128 elems (cheap)
    float nrm = 0.f;
    #pragma unroll 8
    for (int i = 0; i < AD_; ++i) nrm += vw[i] * vw[i];
    nrm = sqrtf(nrm);
    if (b == 0) veff[a] = vg * vw[a] / nrm;

    float acc = bias[a];
    const float* qrow = query + (size_t)b * KD_;
    #pragma unroll 8
    for (int d = 0; d < KD_; ++d) acc += qrow[d] * wq[(size_t)d * AD_ + a];
    qb[b * AD_ + a] = acc;
}

// ---------------------------------------------------------------------------
// Kernel 0b: pre-swizzle W (f32 [512][128]) into bf16 WMMA B-fragments.
// Dest layout: frag f = ks*8+nt ; element (f*32 + lane)*16 + i,
// where lane = (k%32)/16*16 + a%16, i = k%16, nt = a/16, ks = k/32.
// grid = 1024 (mat[2] x k[512]), 128 threads (one per a).
// ---------------------------------------------------------------------------
__global__ __launch_bounds__(128) void mocha_prep_wfrag(
    const float* __restrict__ wk_m, const float* __restrict__ wk_c,
    __bf16* __restrict__ wfm, __bf16* __restrict__ wfc)
{
    const int mat = blockIdx.x >> 9;
    const int k   = blockIdx.x & 511;
    const int a   = threadIdx.x;

    const float* W = mat ? wk_c : wk_m;
    __bf16*    dst = mat ? wfc  : wfm;

    const float x = W[(size_t)k * AD_ + a];
    const int ks   = k >> 5;
    const int kl   = k & 31;
    const int half = kl >> 4;
    const int i    = kl & 15;
    const int nt   = a >> 4;
    const int n    = a & 15;
    const int lane = half * 16 + n;
    dst[(((size_t)(ks * 8 + nt)) * 32 + lane) * 16 + i] = (__bf16)x;
}

// ---------------------------------------------------------------------------
// Kernel 1: fused dual-energy GEMM + tanh/dot epilogue.
// grid = 1024 blocks of 256 threads (8 waves); each wave owns 16 rows,
// computes k = key@W for BOTH weight matrices (key read once), then
// e = sum_a tanh(k + qb) * veff + (vb + r), masked.
// 16 x v_wmma_f32_16x16x32_bf16 per K-step, 16 K-steps.
// ---------------------------------------------------------------------------
__global__ __launch_bounds__(256) void mocha_energy_gemm(
    const float* __restrict__ key, const int* __restrict__ mask,
    const __bf16* __restrict__ wfm, const __bf16* __restrict__ wfc,
    const float* __restrict__ qbm, const float* __restrict__ qbc,
    const float* __restrict__ veffm, const float* __restrict__ veffc,
    const float* __restrict__ vb_m, const float* __restrict__ r_m,
    const float* __restrict__ vb_c, const float* __restrict__ r_c,
    float* __restrict__ em, float* __restrict__ ec)
{
    const int tid   = threadIdx.x;
    const int lane  = tid & 31;
    const int wave  = tid >> 5;
    const int row0  = blockIdx.x * 128 + wave * 16;
    const int b     = row0 >> 12;          // T = 4096 rows per batch
    const int lid   = lane & 15;
    const int upper = lane >> 4;

    const float sm = vb_m[0] + r_m[0];
    const float sc = vb_c[0] + r_c[0];

    const float* krow = key + (size_t)(row0 + lid) * KD_;

    v8f accm[8] = {};
    v8f accc[8] = {};

    for (int ks = 0; ks < 16; ++ks) {
        // A fragment (16x32 bf16): lanes 0-15 hold K {0..15},
        // lanes 16-31 hold K {16..31} of this 32-wide K step.
        const int kb = ks * 32 + (upper ? 16 : 0);
        f4 x0 = *(const f4*)(krow + kb);
        f4 x1 = *(const f4*)(krow + kb + 4);
        f4 x2 = *(const f4*)(krow + kb + 8);
        f4 x3 = *(const f4*)(krow + kb + 12);
        v16bf a;
        a[0]=(__bf16)x0.x; a[1]=(__bf16)x0.y; a[2]=(__bf16)x0.z; a[3]=(__bf16)x0.w;
        a[4]=(__bf16)x1.x; a[5]=(__bf16)x1.y; a[6]=(__bf16)x1.z; a[7]=(__bf16)x1.w;
        a[8]=(__bf16)x2.x; a[9]=(__bf16)x2.y; a[10]=(__bf16)x2.z; a[11]=(__bf16)x2.w;
        a[12]=(__bf16)x3.x; a[13]=(__bf16)x3.y; a[14]=(__bf16)x3.z; a[15]=(__bf16)x3.w;

        const __bf16* pm = wfm + ((size_t)(ks * 8) * 32 + lane) * 16;
        const __bf16* pc = wfc + ((size_t)(ks * 8) * 32 + lane) * 16;
        #pragma unroll
        for (int nt = 0; nt < 8; ++nt) {
            v16bf bm = *(const v16bf*)(pm + (size_t)nt * 32 * 16);
            accm[nt] = __builtin_amdgcn_wmma_f32_16x16x32_bf16(
                false, a, false, bm, (short)0, accm[nt], false, false);
            v16bf bc = *(const v16bf*)(pc + (size_t)nt * 32 * 16);
            accc[nt] = __builtin_amdgcn_wmma_f32_16x16x32_bf16(
                false, a, false, bc, (short)0, accc[nt], false, false);
        }
    }

    // Epilogue: C reg r, lane l holds C[M = r + 8*(l>>4)][N = nt*16 + (l&15)].
    float rm[8] = {0,0,0,0,0,0,0,0};
    float rc[8] = {0,0,0,0,0,0,0,0};
    #pragma unroll
    for (int nt = 0; nt < 8; ++nt) {
        const int aidx = nt * 16 + lid;
        const float qm = qbm[b * AD_ + aidx];
        const float vm = veffm[aidx];
        const float qc = qbc[b * AD_ + aidx];
        const float vc = veffc[aidx];
        #pragma unroll
        for (int r = 0; r < 8; ++r) {
            rm[r] += tanhf(accm[nt][r] + qm) * vm;
            rc[r] += tanhf(accc[nt][r] + qc) * vc;
        }
    }
    // reduce across the 16 lanes of each half
    #pragma unroll
    for (int r = 0; r < 8; ++r) {
        #pragma unroll
        for (int m = 1; m < 16; m <<= 1) {
            rm[r] += __shfl_xor(rm[r], m, 16);
            rc[r] += __shfl_xor(rc[r], m, 16);
        }
    }
    if (lid == 0) {
        #pragma unroll
        for (int r = 0; r < 8; ++r) {
            const int row = row0 + r + 8 * upper;
            float e1 = rm[r] + sm;
            float e2 = rc[r] + sc;
            if (mask[row] == 0) { e1 = NEG_INF; e2 = NEG_INF; }
            em[row] = e1;
            ec[row] = e2;
        }
    }
}

// ---------------------------------------------------------------------------
// Kernel 2: per-batch-row scan. grid = 32 blocks x 256 threads,
// each thread owns 16 consecutive t. Computes alpha (monotonic) and beta
// (chunkwise, window=8), writes alpha to d_out and beta to ws.
// Note: cumsum(aw_prev/cumprod) == 1 for the one-hot prev, so
// alpha[t] = p[t] * prod_{s<t}(1-p[s]).
// ---------------------------------------------------------------------------
__global__ __launch_bounds__(256) void mocha_scan(
    const float* __restrict__ em, const float* __restrict__ ec,
    const float* __restrict__ noise,
    float* __restrict__ alpha_out,   // d_out + 16384
    float* __restrict__ beta)
{
    __shared__ float sA[T_LEN];
    __shared__ float sB[T_LEN];
    __shared__ float swv[8];

    const int b   = blockIdx.x;
    const int tid = threadIdx.x;
    const int t0  = tid * 16;
    const int lane = tid & 31;
    const int wid  = tid >> 5;
    const size_t base = (size_t)b * T_LEN;

    float p[16], lp[16];
    float lsum = 0.f;
    #pragma unroll
    for (int i = 0; i < 16; ++i) {
        const int t = t0 + i;
        const float e = em[base + t] + noise[base + t];
        const float pi = 1.f / (1.f + expf(-e));
        p[i] = pi;
        float om = 1.f - pi;
        om = fminf(fmaxf(om, 1e-10f), 1.f);
        lp[i] = logf(om);
        lsum += lp[i];
    }
    // block-wide exclusive scan of per-thread chunk sums
    float inc = lsum;
    #pragma unroll
    for (int off = 1; off < 32; off <<= 1) {
        const float nb = __shfl_up(inc, off, 32);
        if (lane >= off) inc += nb;
    }
    if (lane == 31) swv[wid] = inc;
    __syncthreads();
    if (tid == 0) {
        float run = 0.f;
        #pragma unroll
        for (int w = 0; w < 8; ++w) { const float t = swv[w]; swv[w] = run; run += t; }
    }
    __syncthreads();
    float run = (inc - lsum) + swv[wid];  // exclusive prefix at t0

    #pragma unroll
    for (int i = 0; i < 16; ++i) {
        const float al = p[i] * expf(run);  // p * cumprod_{s<t}(1-p)
        run += lp[i];
        sA[t0 + i] = al;
        alpha_out[base + t0 + i] = al;
    }

    // chunk energy softmax (row max reduction)
    float ecv[16];
    float lmax = NEG_INF;
    #pragma unroll
    for (int i = 0; i < 16; ++i) {
        ecv[i] = ec[base + t0 + i];
        lmax = fmaxf(lmax, ecv[i]);
    }
    #pragma unroll
    for (int m = 1; m < 32; m <<= 1) lmax = fmaxf(lmax, __shfl_xor(lmax, m, 32));
    __syncthreads();            // swv free again
    if (lane == 0) swv[wid] = lmax;
    __syncthreads();
    float bmax = swv[0];
    #pragma unroll
    for (int w = 1; w < 8; ++w) bmax = fmaxf(bmax, swv[w]);

    #pragma unroll
    for (int i = 0; i < 16; ++i)
        sB[t0 + i] = fmaxf(expf(ecv[i] - bmax), 1e-5f);
    __syncthreads();

    // ratio[t] = alpha[t] / movsum(sexp, back=7, fwd=0)[t]
    float ratio[16];
    #pragma unroll
    for (int i = 0; i < 16; ++i) {
        const int t = t0 + i;
        float den = 0.f;
        const int lo = (t >= 7) ? (t - 7) : 0;
        for (int s = lo; s <= t; ++s) den += sB[s];
        ratio[i] = sA[t] / den;
    }
    #pragma unroll
    for (int i = 0; i < 16; ++i) sA[t0 + i] = ratio[i];
    __syncthreads();

    // beta[t] = sexp[t] * movsum(ratio, back=0, fwd=7)[t]
    #pragma unroll
    for (int i = 0; i < 16; ++i) {
        const int t = t0 + i;
        float s = 0.f;
        #pragma unroll
        for (int off = 0; off < 8; ++off) {
            const int u = t + off;
            if (u < T_LEN) s += sA[u];
        }
        beta[base + t] = sB[t] * s;
    }
}

// ---------------------------------------------------------------------------
// Kernel 3a: partial context cv_part[b][seg][v] = sum_{t in seg} beta*value.
// grid = B * NSEG blocks x 256 threads; each thread owns 2 consecutive
// columns (float2 loads on the 256 MB `value` stream).
// beta segment is staged into LDS via the CDNA5 async-to-LDS path
// (ASYNCcnt-tracked), then broadcast-read from LDS.
// ---------------------------------------------------------------------------
__global__ __launch_bounds__(256) void mocha_ctx_partial(
    const float* __restrict__ value, const float* __restrict__ beta,
    float* __restrict__ part)
{
    __shared__ float sb[T_LEN / NSEG];   // 256 beta values

    const int x    = blockIdx.x;         // b*NSEG + seg
    const int b    = x / NSEG;
    const int seg  = x % NSEG;
    const int tid  = threadIdx.x;
    const int v2   = tid * 2;            // two consecutive columns
    const int tseg = T_LEN / NSEG;       // 256
    const int t0   = seg * tseg;

    // async-copy beta[t0 .. t0+255] into LDS (one b32 per lane)
    {
        const float* gsrc = beta + (size_t)b * T_LEN + t0 + tid;
        const unsigned lds_off = (unsigned)(uintptr_t)(&sb[tid]);
        const unsigned long long ga = (unsigned long long)(uintptr_t)gsrc;
        asm volatile("global_load_async_to_lds_b32 %0, %1, off"
                     :: "v"(lds_off), "v"(ga) : "memory");
        asm volatile("s_wait_asynccnt 0" ::: "memory");
    }
    __syncthreads();

    float2 acc = make_float2(0.f, 0.f);
    const float* vp = value + ((size_t)b * T_LEN + t0) * KD_ + v2;
    #pragma unroll 4
    for (int i = 0; i < tseg; ++i) {
        const float2 w = *(const float2*)(vp + (size_t)i * KD_);
        const float bw = sb[i];
        acc.x += bw * w.x;
        acc.y += bw * w.y;
    }
    float* pp = part + ((size_t)(b * NSEG + seg)) * KD_ + v2;
    pp[0] = acc.x;
    pp[1] = acc.y;
}

// ---------------------------------------------------------------------------
// Kernel 3b: reduce partials into cv (front of d_out).
// grid = 32 blocks x 512 threads.
// ---------------------------------------------------------------------------
__global__ __launch_bounds__(512) void mocha_ctx_reduce(
    const float* __restrict__ part, float* __restrict__ cv)
{
    const int b = blockIdx.x;
    const int v = threadIdx.x;
    float s = 0.f;
    #pragma unroll
    for (int seg = 0; seg < NSEG; ++seg)
        s += part[((size_t)(b * NSEG + seg)) * KD_ + v];
    cv[(size_t)b * KD_ + v] = s;
}

// ---------------------------------------------------------------------------
// Host launcher.
// ---------------------------------------------------------------------------
extern "C" void kernel_launch(void* const* d_in, const int* in_sizes, int n_in,
                              void* d_out, int out_size, void* d_ws, size_t ws_size,
                              hipStream_t stream) {
    const float* key   = (const float*)d_in[0];
    const float* value = (const float*)d_in[1];
    const float* query = (const float*)d_in[2];
    const int*   mask  = (const int*)  d_in[3];
    const float* noise = (const float*)d_in[4];
    /* window = d_in[5] (== 8, hardcoded) */
    const float* wk_m = (const float*)d_in[6];
    const float* wq_m = (const float*)d_in[7];
    const float* b_m  = (const float*)d_in[8];
    const float* vw_m = (const float*)d_in[9];
    const float* vg_m = (const float*)d_in[10];
    const float* vb_m = (const float*)d_in[11];
    const float* r_m  = (const float*)d_in[12];
    const float* wk_c = (const float*)d_in[13];
    const float* wq_c = (const float*)d_in[14];
    const float* b_c  = (const float*)d_in[15];
    const float* vw_c = (const float*)d_in[16];
    const float* vg_c = (const float*)d_in[17];
    const float* vb_c = (const float*)d_in[18];
    const float* r_c  = (const float*)d_in[19];

    // workspace layout (bytes, all 512B-aligned chunks)
    char* ws = (char*)d_ws;
    size_t off = 0;
    float* em    = (float*)(ws + off); off += (size_t)B_SZ * T_LEN * 4;        // 512 KB
    float* ec    = (float*)(ws + off); off += (size_t)B_SZ * T_LEN * 4;        // 512 KB
    float* beta  = (float*)(ws + off); off += (size_t)B_SZ * T_LEN * 4;        // 512 KB
    float* qbm   = (float*)(ws + off); off += (size_t)B_SZ * AD_ * 4;          // 16 KB
    float* qbc   = (float*)(ws + off); off += (size_t)B_SZ * AD_ * 4;          // 16 KB
    float* veffm = (float*)(ws + off); off += AD_ * 4;
    float* veffc = (float*)(ws + off); off += AD_ * 4;
    float* part  = (float*)(ws + off); off += (size_t)B_SZ * NSEG * KD_ * 4;   // 1 MB
    __bf16* wfm  = (__bf16*)(ws + off); off += (size_t)KD_ * AD_ * 2;          // 128 KB
    __bf16* wfc  = (__bf16*)(ws + off); off += (size_t)KD_ * AD_ * 2;          // 128 KB

    float* cv_out    = (float*)d_out;                 // [32][512]
    float* alpha_out = (float*)d_out + B_SZ * KD_;    // [32][4096]

    mocha_prep_vq<<<64, 128, 0, stream>>>(
        query, wq_m, b_m, vw_m, vg_m, wq_c, b_c, vw_c, vg_c,
        qbm, qbc, veffm, veffc);

    mocha_prep_wfrag<<<2 * KD_, 128, 0, stream>>>(wk_m, wk_c, wfm, wfc);

    mocha_energy_gemm<<<(B_SZ * T_LEN) / 128, 256, 0, stream>>>(
        key, mask, wfm, wfc, qbm, qbc, veffm, veffc,
        vb_m, r_m, vb_c, r_c, em, ec);

    mocha_scan<<<B_SZ, 256, 0, stream>>>(em, ec, noise, alpha_out, beta);

    mocha_ctx_partial<<<B_SZ * NSEG, 256, 0, stream>>>(value, beta, part);

    mocha_ctx_reduce<<<B_SZ, 512, 0, stream>>>(part, cv_out);
}
